// ConvPixelToCapsules_52862457479577
// MI455X (gfx1250) — compile-verified, matrix operand order checked by reading, not run on hardware
//
#include <hip/hip_runtime.h>
#include <hip/hip_bf16.h>

// CDNA5 / gfx1250 fused capsule-routing kernel.
// Phase 1: per-(b,h,w) 256x256xK(=96) GEMM via v_wmma_f32_16x16x32_bf16 -> votes in LDS.
// Phase 2: 3 dynamic-routing iterations entirely in LDS/registers.
// HBM traffic: x (6.5MB) read + activation (0.6MB) written. votes (151MB) never leaves the WGP.

typedef __bf16 v16bf __attribute__((ext_vector_type(16)));
typedef float  v8f   __attribute__((ext_vector_type(8)));

#define CO 32
#define NO 8
#define CI 256
#define KK 81          // 9x9 kernel taps
#define HI 20
#define WI 20
#define HO 6
#define WO 6
#define VSTRIDE 258    // bf16 halves per votes row: 256 + 2 pad -> 129 dwords (odd) => conflict-free column reads
#define RSTRIDE 34     // bf16 halves per route row: 32 + 2 pad  -> 17 dwords (odd)
#define SMEM_BYTES (CI * VSTRIDE * 2 + CI * RSTRIDE * 2 + CO * NO * 4 + CO * NO * 4)

// ---------------------------------------------------------------------------
// Prep: swizzle W (256 x 81, f32) into WMMA B-fragment order, bf16, K padded to 96.
// Layout: Bfrag[ntile(16)][kblock(3)][lane(32)][16 halves].
// 16-bit B 32x16 layout: lane<16 -> N=lane,   K = {2v,2v+1} (v<4) / {16+2(v-4),..} per dword
//                        lane>=16 -> N=lane-16, same K pattern + 8.
// ---------------------------------------------------------------------------
__global__ void caps_prep_B(const float* __restrict__ W, __bf16* __restrict__ Bmat) {
  const int nt    = blockIdx.x;   // 0..15
  const int kb    = blockIdx.y;   // 0..2
  const int lane  = threadIdx.x;  // 0..31
  const int n     = lane & 15;
  const int khalf = lane >> 4;
  const int ncol  = nt * 16 + n;  // global output channel = c*8 + no
  v16bf frag;
#pragma unroll
  for (int v = 0; v < 8; ++v) {
    const int k0 = (v < 4 ? 2 * v : 16 + 2 * (v - 4)) + 8 * khalf + 32 * kb;
#pragma unroll
    for (int e = 0; e < 2; ++e) {
      const int k = k0 + e;
      const float val = (k < KK) ? W[ncol * KK + k] : 0.0f;
      frag[2 * v + e] = (__bf16)val;
    }
  }
  *(v16bf*)(Bmat + ((size_t)(nt * 3 + kb) * 32 + lane) * 16) = frag;
}

// ---------------------------------------------------------------------------
// Fused GEMM + routing. One workgroup (256 threads = 8 waves) per (b, h, w).
// ---------------------------------------------------------------------------
__global__ __launch_bounds__(256) void caps_route(
    const float* __restrict__ x, const float* __restrict__ bias,
    const __bf16* __restrict__ Bmat, float* __restrict__ out) {
  extern __shared__ char smem[];
  __bf16* votes_l = (__bf16*)smem;                              // [CI][VSTRIDE]
  __bf16* route_l = (__bf16*)(smem + CI * VSTRIDE * 2);         // [CI][RSTRIDE]
  float*  pre_l   = (float*)(smem + CI * VSTRIDE * 2 + CI * RSTRIDE * 2);  // [256]
  float*  act_l   = pre_l + CO * NO;                            // [256]

  const int w = blockIdx.x, h = blockIdx.y, b = blockIdx.z;
  const int t     = threadIdx.x;
  const int lane  = t & 31;
  const int wv    = t >> 5;     // wave 0..7 -> rows [32*wv, 32*wv+32)
  const int mr    = lane & 15;
  const int khalf = lane >> 4;

  // ---- Phase 1a: gather A fragments (x patch rows, bf16, WMMA A-layout, K padded to 96)
  v16bf afrag[2][3];
#pragma unroll
  for (int mt = 0; mt < 2; ++mt) {
    const int row = wv * 32 + mt * 16 + mr;   // ci index
    const float* xrow = x + ((size_t)(b * CI + row)) * (HI * WI) + (2 * h) * WI + 2 * w;
#pragma unroll
    for (int kb = 0; kb < 3; ++kb) {
      v16bf a;
#pragma unroll
      for (int v = 0; v < 8; ++v) {
        const int k0 = (v < 4 ? 2 * v : 16 + 2 * (v - 4)) + 8 * khalf + 32 * kb;
#pragma unroll
        for (int e = 0; e < 2; ++e) {
          const int k = k0 + e;
          float val = 0.0f;
          if (k < KK) {
            const int kh = k / 9;
            const int kw = k - 9 * kh;
            val = xrow[kh * WI + kw];
          }
          a[2 * v + e] = (__bf16)val;
        }
      }
      afrag[mt][kb] = a;
    }
  }

  // ---- Phase 1b: 16 n-tiles x 2 m-tiles x 3 K-steps of WMMA; scatter f32->bf16 into LDS votes
  for (int nt = 0; nt < 16; ++nt) {
    const v16bf b0 = *(const v16bf*)(Bmat + ((size_t)(nt * 3 + 0) * 32 + lane) * 16);
    const v16bf b1 = *(const v16bf*)(Bmat + ((size_t)(nt * 3 + 1) * 32 + lane) * 16);
    const v16bf b2 = *(const v16bf*)(Bmat + ((size_t)(nt * 3 + 2) * 32 + lane) * 16);
#pragma unroll
    for (int mt = 0; mt < 2; ++mt) {
      v8f acc = {0.f, 0.f, 0.f, 0.f, 0.f, 0.f, 0.f, 0.f};
      acc = __builtin_amdgcn_wmma_f32_16x16x32_bf16(false, afrag[mt][0], false, b0, (short)0, acc, false, false);
      acc = __builtin_amdgcn_wmma_f32_16x16x32_bf16(false, afrag[mt][1], false, b1, (short)0, acc, false, false);
      acc = __builtin_amdgcn_wmma_f32_16x16x32_bf16(false, afrag[mt][2], false, b2, (short)0, acc, false, false);
      // D layout: VGPR r holds (M = r + 8*khalf, N = lane&15)
      const int rbase = wv * 32 + mt * 16 + 8 * khalf;
      const int col   = nt * 16 + (lane & 15);
#pragma unroll
      for (int r = 0; r < 8; ++r)
        votes_l[(rbase + r) * VSTRIDE + col] = (__bf16)acc[r];
    }
  }
  __syncthreads();

  // ---- Phase 2: dynamic routing (3 iterations), logits/route in registers
  const int   cT = t >> 3;                  // (c,n) role of this thread
  const int   nT = t & 7;
  const float biasv = bias[t];              // bias is (32,8,1,1) flat = 256
  float logits_[CO];
#pragma unroll
  for (int c = 0; c < CO; ++c) logits_[c] = 0.0f;

  for (int it = 0; it < 3; ++it) {
    // softmax over co for row i = t (registers), store route to LDS
    float m = logits_[0];
#pragma unroll
    for (int c = 1; c < CO; ++c) m = fmaxf(m, logits_[c]);
    float rt[CO];
    float s = 0.0f;
#pragma unroll
    for (int c = 0; c < CO; ++c) { rt[c] = __expf(logits_[c] - m); s += rt[c]; }
    const float inv_s = 1.0f / s;
#pragma unroll
    for (int c = 0; c < CO; ++c) route_l[t * RSTRIDE + c] = (__bf16)(rt[c] * inv_s);
    __syncthreads();

    // preactivate[c,n] = bias + sum_i route[i,c] * votes[i,c*8+n]   (thread = (c,n))
    float acc = biasv;
#pragma unroll 4
    for (int i = 0; i < CI; ++i)
      acc += (float)route_l[i * RSTRIDE + cT] * (float)votes_l[i * VSTRIDE + t];
    pre_l[t] = acc;
    __syncthreads();

    // squash over no: act = pre * ||pre|| / (1 + ||pre||^2)
    float ss = 0.0f;
#pragma unroll
    for (int n = 0; n < NO; ++n) { const float p = pre_l[cT * NO + n]; ss += p * p; }
    const float norm  = __fsqrt_rn(ss);
    const float scale = norm / (1.0f + ss);
    const float act   = acc * scale;
    act_l[t] = act;
    __syncthreads();

    if (it == 2) {
      // activation -> out[b, c, n, h, w]
      out[(((size_t)(b * CO + cT) * NO + nT) * HO + h) * WO + w] = act;
    } else {
      // distances[i,c] = sum_n votes[i,c*8+n] * act[c,n]; logits += distances (thread = i)
#pragma unroll
      for (int c = 0; c < CO; ++c) {
        float d = 0.0f;
#pragma unroll
        for (int n = 0; n < NO; ++n)
          d += (float)votes_l[t * VSTRIDE + c * NO + n] * act_l[c * NO + n];
        logits_[c] += d;
      }
      __syncthreads();
    }
  }
}

extern "C" void kernel_launch(void* const* d_in, const int* in_sizes, int n_in,
                              void* d_out, int out_size, void* d_ws, size_t ws_size,
                              hipStream_t stream) {
  const float* x    = (const float*)d_in[0];   // (16, 256, 1, 20, 20)
  const float* W    = (const float*)d_in[1];   // (256, 1, 9, 9)
  const float* bias = (const float*)d_in[2];   // (32, 8, 1, 1)
  float* out = (float*)d_out;                  // (16, 32, 8, 6, 6)
  __bf16* Bmat = (__bf16*)d_ws;                // 16*3*32*16 bf16 = 48KB

  caps_prep_B<<<dim3(16, 3), 32, 0, stream>>>(W, Bmat);
  caps_route<<<dim3(WO, HO, 16), 256, SMEM_BYTES, stream>>>(x, bias, Bmat, out);
}
